// ContrastiveAffinityLossWithMemory_8340826489556
// MI455X (gfx1250) — compile-verified, matrix operand order checked by reading, not run on hardware
//
#include <hip/hip_runtime.h>
#include <hip/hip_bf16.h>

// ---- problem constants (match reference) ----
#define B_SZ 4096
#define D_SZ 512
#define C_SZ 4096
#define DD   384            // int(D * 0.75)
#define MARGIN_F   4.0f
#define ALPHA_F    0.7f
#define DECAY_F    0.01f
#define CURTIME_F  1.0f
#define EPS_F      1e-12f
#define BG_SIM_F       0.2f
#define BG_OTHER_SIM_F 0.01f

typedef __attribute__((ext_vector_type(16))) _Float16 v16h;
typedef __attribute__((ext_vector_type(8)))  float    v8f;

#define TILE       128
#define LDS_K      32
#define LDS_STRIDE 40      // 32 + 8 halves pad -> 80B row stride, 16B aligned
#define NKSTEP     (DD / LDS_K)   // 12
#define BUF_ELEMS  (TILE * LDS_STRIDE)   // halves per buffer

// ---------------------------------------------------------------------------
// CDNA5 async global->LDS copy (16B per lane), tracked by ASYNCcnt
// LDS aperture keeps the byte offset in addr[31:0] (ISA 10.2), so the low
// 32 bits of a generic pointer to __shared__ are the LDS address.
// ---------------------------------------------------------------------------
__device__ __forceinline__ void async_copy_b128(void* lds_dst, const void* gsrc) {
  unsigned lds = (unsigned)(uintptr_t)lds_dst;
  asm volatile("global_load_async_to_lds_b128 %0, %1, off"
               :: "v"(lds), "v"(gsrc) : "memory");
}
__device__ __forceinline__ void wait_async_le4() {
  asm volatile("s_wait_asynccnt 0x4" ::: "memory");
}
__device__ __forceinline__ void wait_async_0() {
  asm volatile("s_wait_asynccnt 0x0" ::: "memory");
}

// ---------------------------------------------------------------------------
// 0) zero the scalar accumulators (pair_sum, mem_sum, n_init, n_valid)
// ---------------------------------------------------------------------------
__global__ void k_init(float* acc) {
  if (threadIdx.x == 0) {
    acc[0] = 0.0f;           // pair loss sum
    acc[1] = 0.0f;           // mem loss sum
    ((int*)acc)[2] = 0;      // n_init
    ((int*)acc)[3] = 0;      // n_valid
  }
}

// ---------------------------------------------------------------------------
// 1) per-class sequential EMA memory update + normalize -> mh (f16), w, init
// ---------------------------------------------------------------------------
__global__ __launch_bounds__(128) void k_update_mem(
    const int*   __restrict__ y_true,
    const float* __restrict__ y_pred,
    const float* __restrict__ mem_emb,
    const float* __restrict__ mem_ts,
    const int*   __restrict__ mem_init,
    _Float16*    __restrict__ mh,
    float*       __restrict__ wv,
    int*         __restrict__ init_new,
    float*       __restrict__ acc)
{
  __shared__ int   s_match[256];
  __shared__ int   s_cnt;
  __shared__ float s_red[128];

  const int tid = threadIdx.x;
  const int c   = blockIdx.x;

  if (tid == 0) s_cnt = 0;
  __syncthreads();

  // parallel strided scan of labels; collect matching sample indices
  for (int i = tid; i < B_SZ; i += 128) {
    if (y_true[i] == c) {
      int slot = atomicAdd(&s_cnt, 1);
      if (slot < 256) s_match[slot] = i;
    }
  }
  __syncthreads();
  int cnt = s_cnt; if (cnt > 256) cnt = 256;

  // restore sequential order (EMA is order-dependent): tiny insertion sort
  if (tid == 0 && cnt > 1) {
    for (int a = 1; a < cnt; ++a) {
      int v = s_match[a]; int b = a - 1;
      while (b >= 0 && s_match[b] > v) { s_match[b + 1] = s_match[b]; --b; }
      s_match[b + 1] = v;
    }
  }
  __syncthreads();

  float e0 = mem_emb[(size_t)c * DD + tid];
  float e1 = mem_emb[(size_t)c * DD + tid + 128];
  float e2 = mem_emb[(size_t)c * DD + tid + 256];
  bool  init = (mem_init[c] != 0);
  float ts   = mem_ts[c];

  for (int j = 0; j < cnt; ++j) {
    const float* pr = y_pred + (size_t)s_match[j] * D_SZ;
    float v0 = pr[tid], v1 = pr[tid + 128], v2 = pr[tid + 256];
    if (init) {
      e0 = ALPHA_F * v0 + (1.0f - ALPHA_F) * e0;
      e1 = ALPHA_F * v1 + (1.0f - ALPHA_F) * e1;
      e2 = ALPHA_F * v2 + (1.0f - ALPHA_F) * e2;
    } else {
      e0 = v0; e1 = v1; e2 = v2;
    }
    init = true; ts = CURTIME_F;
  }

  // L2 norm across the 384-dim row
  s_red[tid] = e0 * e0 + e1 * e1 + e2 * e2;
  __syncthreads();
  for (int s = 64; s > 0; s >>= 1) {
    if (tid < s) s_red[tid] += s_red[tid + s];
    __syncthreads();
  }
  float inv = init ? (1.0f / fmaxf(sqrtf(s_red[0]), EPS_F)) : 0.0f;

  mh[(size_t)c * DD + tid]       = (_Float16)(e0 * inv);
  mh[(size_t)c * DD + tid + 128] = (_Float16)(e1 * inv);
  mh[(size_t)c * DD + tid + 256] = (_Float16)(e2 * inv);

  if (tid == 0) {
    wv[c]       = init ? expf(-DECAY_F * (CURTIME_F - ts)) : 0.0f;
    init_new[c] = init ? 1 : 0;
    if (init) atomicAdd(((int*)acc) + 2, 1);
  }
}

// ---------------------------------------------------------------------------
// 2) normalize y_pred[:, :384] -> xh (f16); count valid samples
// ---------------------------------------------------------------------------
__global__ __launch_bounds__(128) void k_norm_x(
    const int*   __restrict__ y_true,
    const float* __restrict__ y_pred,
    _Float16*    __restrict__ xh,
    float*       __restrict__ acc)
{
  __shared__ float s_red[128];
  const int tid = threadIdx.x;
  const int i   = blockIdx.x;
  const float* pr = y_pred + (size_t)i * D_SZ;
  float v0 = pr[tid], v1 = pr[tid + 128], v2 = pr[tid + 256];
  s_red[tid] = v0 * v0 + v1 * v1 + v2 * v2;
  __syncthreads();
  for (int s = 64; s > 0; s >>= 1) {
    if (tid < s) s_red[tid] += s_red[tid + s];
    __syncthreads();
  }
  float inv = 1.0f / fmaxf(sqrtf(s_red[0]), EPS_F);
  xh[(size_t)i * DD + tid]       = (_Float16)(v0 * inv);
  xh[(size_t)i * DD + tid + 128] = (_Float16)(v1 * inv);
  xh[(size_t)i * DD + tid + 256] = (_Float16)(v2 * inv);
  if (tid == 0) {
    int yt = y_true[i];
    if (yt >= 0 && yt < C_SZ) atomicAdd(((int*)acc) + 3, 1);
  }
}

// ---------------------------------------------------------------------------
// shared 128x128xK GEMM core on v_wmma_f32_16x16x32_f16
//   block = 256 threads (8 waves); wave w owns rows [w*16, w*16+16) x 128 cols
//   contiguous double-buffered LDS (buffer select = integer element offset so
//   InferAddressSpaces keeps ds_load), staged via global_load_async_to_lds_b128
// ---------------------------------------------------------------------------
__device__ __forceinline__ void stage_chunk(
    const _Float16* __restrict__ Ag, const _Float16* __restrict__ Bg,
    int rowBaseA, int rowBaseB, int kk,
    _Float16* Ash, _Float16* Bsh, int boff, int tid)
{
  #pragma unroll
  for (int it = 0; it < 2; ++it) {
    int idx = tid + it * 256;        // 0..511 -> 512 x 16B chunks per array
    int r   = idx >> 2;              // 0..127
    int k8  = (idx & 3) * 8;         // 0,8,16,24 halves
    async_copy_b128(Ash + boff + r * LDS_STRIDE + k8,
                    Ag + (size_t)(rowBaseA + r) * DD + kk + k8);
    async_copy_b128(Bsh + boff + r * LDS_STRIDE + k8,
                    Bg + (size_t)(rowBaseB + r) * DD + kk + k8);
  }
}

__device__ __forceinline__ void mm128(
    const _Float16* __restrict__ Ag, const _Float16* __restrict__ Bg,
    int rowBaseA, int rowBaseB,
    _Float16* Ash, _Float16* Bsh,     // each 2*BUF_ELEMS halves (double buffer)
    v8f (&acc8)[8])
{
  const int tid  = threadIdx.x;
  const int lane = tid & 31;
  const int wave = tid >> 5;
  const int wrow = wave * 16;
  const int kb   = (lane & 16) ? 8 : 0;   // 16-bit A/B fragment K-offset per ISA layout

  // prime buffer 0 (4 async ops per wave)
  stage_chunk(Ag, Bg, rowBaseA, rowBaseB, 0, Ash, Bsh, 0, tid);

  for (int kc = 0; kc < NKSTEP; ++kc) {
    const int cboff = (kc & 1) * BUF_ELEMS;          // current buffer offset
    if (kc + 1 < NKSTEP) {
      // overlap: issue next chunk into the other buffer, wait for previous 4
      stage_chunk(Ag, Bg, rowBaseA, rowBaseB, (kc + 1) * LDS_K,
                  Ash, Bsh, BUF_ELEMS - cboff, tid);
      wait_async_le4();
    } else {
      wait_async_0();
    }
    __syncthreads();   // all waves' async copies for the current buffer visible

    // A fragment: lane L holds row (L&15); K = {0..7,16..23} (+8 for lanes 16..31)
    v16h af;
    const _Float16* ap = Ash + cboff + (wrow + (lane & 15)) * LDS_STRIDE;
    #pragma unroll
    for (int i = 0; i < 16; ++i) {
      int k = ((i < 8) ? i : i + 8) + kb;
      af[i] = ap[k];
    }

    // preload all 8 B fragments, then burst the 8 WMMAs
    v16h bf[8];
    #pragma unroll
    for (int f = 0; f < 8; ++f) {
      const _Float16* bp = Bsh + cboff + (f * 16 + (lane & 15)) * LDS_STRIDE;
      #pragma unroll
      for (int i = 0; i < 16; ++i) {
        int k = ((i < 8) ? i : i + 8) + kb;
        bf[f][i] = bp[k];
      }
    }
    #pragma unroll
    for (int f = 0; f < 8; ++f) {
      acc8[f] = __builtin_amdgcn_wmma_f32_16x16x32_f16(
          false, af, false, bf[f], (short)0, acc8[f], false, false);
    }
    __syncthreads();   // done reading current buffer before it is overwritten
  }
}

// ---------------------------------------------------------------------------
// 3) pair loss: cos = x @ x^T over upper-triangular 128x128 tile pairs
// ---------------------------------------------------------------------------
__global__ __launch_bounds__(256) void k_pair_loss(
    const _Float16* __restrict__ xh,
    const int*      __restrict__ y_true,
    float*          __restrict__ acc)
{
  __shared__ _Float16 Ash[2 * BUF_ELEMS];
  __shared__ _Float16 Bsh[2 * BUF_ELEMS];
  __shared__ float    s_red[256];

  // decode triangular tile pair (bi <= bj) from linear block id
  int t = blockIdx.x, bi = 0;
  while (t >= (32 - bi)) { t -= (32 - bi); ++bi; }
  const int bj = bi + t;
  const int rowBaseA = bi * TILE;
  const int rowBaseB = bj * TILE;

  v8f acc8[8];
  #pragma unroll
  for (int f = 0; f < 8; ++f)
    #pragma unroll
    for (int e = 0; e < 8; ++e) acc8[f][e] = 0.0f;

  mm128(xh, xh, rowBaseA, rowBaseB, Ash, Bsh, acc8);

  const int tid  = threadIdx.x;
  const int lane = tid & 31;
  const int wave = tid >> 5;
  const int rbase = rowBaseA + wave * 16 + ((lane & 16) ? 8 : 0);
  const int cbase = rowBaseB + (lane & 15);

  float lsum = 0.0f;
  #pragma unroll
  for (int f = 0; f < 8; ++f) {
    const int gj  = cbase + f * 16;
    const bool bgj = (y_true[gj] == -1);
    #pragma unroll
    for (int r = 0; r < 8; ++r) {
      const int gi = rbase + r;
      if (gj > gi) {
        float cv   = acc8[f][r];
        float sqd  = fmaxf(2.0f - 2.0f * cv, 0.0f);   // dist^2 on triu
        float dist = sqrtf(sqd);
        bool bgi   = (y_true[gi] == -1);
        float tsim = (bgi && bgj) ? BG_SIM_F : ((bgi != bgj) ? BG_OTHER_SIM_F : 0.0f);
        float md   = fmaxf(MARGIN_F - dist, 0.0f);
        lsum += tsim * sqd + (1.0f - tsim) * md * md;
      }
    }
  }

  s_red[tid] = lsum;
  __syncthreads();
  for (int s = 128; s > 0; s >>= 1) {
    if (tid < s) s_red[tid] += s_red[tid + s];
    __syncthreads();
  }
  if (tid == 0) atomicAdd(&acc[0], s_red[0]);
}

// ---------------------------------------------------------------------------
// 4) memory loss: cos_m = x @ m^T, weighted by lookup gather + decay weights
// ---------------------------------------------------------------------------
__global__ __launch_bounds__(256) void k_mem_loss(
    const _Float16* __restrict__ xh,
    const _Float16* __restrict__ mh,
    const int*      __restrict__ y_true,
    const float*    __restrict__ lookup,
    const float*    __restrict__ wv,
    const int*      __restrict__ init_new,
    float*          __restrict__ acc)
{
  __shared__ _Float16 Ash[2 * BUF_ELEMS];
  __shared__ _Float16 Bsh[2 * BUF_ELEMS];
  __shared__ float    s_red[256];

  const int rowBaseA = blockIdx.y * TILE;   // batch rows
  const int rowBaseB = blockIdx.x * TILE;   // class rows

  v8f acc8[8];
  #pragma unroll
  for (int f = 0; f < 8; ++f)
    #pragma unroll
    for (int e = 0; e < 8; ++e) acc8[f][e] = 0.0f;

  mm128(xh, mh, rowBaseA, rowBaseB, Ash, Bsh, acc8);

  const int tid  = threadIdx.x;
  const int lane = tid & 31;
  const int wave = tid >> 5;
  const int rbase = rowBaseA + wave * 16 + ((lane & 16) ? 8 : 0);
  const int cbase = rowBaseB + (lane & 15);

  // preload per-row labels for this lane's 8 rows
  int yts[8];
  #pragma unroll
  for (int r = 0; r < 8; ++r) yts[r] = y_true[rbase + r];

  float lsum = 0.0f;
  #pragma unroll
  for (int f = 0; f < 8; ++f) {
    const int   cc = cbase + f * 16;
    const int   ic = init_new[cc];
    const float wc = wv[cc];
    if (ic) {
      #pragma unroll
      for (int r = 0; r < 8; ++r) {
        const int yt = yts[r];
        if (yt >= 0 && yt < C_SZ) {
          float tsim = lookup[(size_t)yt * C_SZ + cc];
          float cv   = acc8[f][r];
          float sqd  = fmaxf(2.0f - 2.0f * cv, 0.0f);
          float d2   = fmaxf(sqd, EPS_F);      // dist_m^2
          float dist = sqrtf(d2);
          float md   = fmaxf(MARGIN_F - dist, 0.0f);
          lsum += (tsim * d2 + (1.0f - tsim) * md * md) * wc;
        }
      }
    }
  }

  s_red[tid] = lsum;
  __syncthreads();
  for (int s = 128; s > 0; s >>= 1) {
    if (tid < s) s_red[tid] += s_red[tid + s];
    __syncthreads();
  }
  if (tid == 0) atomicAdd(&acc[1], s_red[0]);
}

// ---------------------------------------------------------------------------
// 5) combine
// ---------------------------------------------------------------------------
__global__ void k_final(const float* __restrict__ acc, float* __restrict__ out) {
  if (threadIdx.x == 0 && blockIdx.x == 0) {
    float pair = acc[0];
    float mem  = acc[1];
    int ninit  = ((const int*)acc)[2]; if (ninit  < 1) ninit  = 1;
    int nvalid = ((const int*)acc)[3]; if (nvalid < 1) nvalid = 1;
    const float n_pairs = (float)(((long long)B_SZ * (B_SZ - 1)) / 2);
    out[0] = 0.7f * (pair / n_pairs) + 0.3f * (mem / ((float)ninit * (float)nvalid));
  }
}

// ---------------------------------------------------------------------------
extern "C" void kernel_launch(void* const* d_in, const int* in_sizes, int n_in,
                              void* d_out, int out_size, void* d_ws, size_t ws_size,
                              hipStream_t stream) {
  (void)in_sizes; (void)n_in; (void)out_size; (void)ws_size;

  const int*   y_true   = (const int*)  d_in[0];
  const float* y_pred   = (const float*)d_in[1];
  const float* lookup   = (const float*)d_in[2];
  const float* mem_emb  = (const float*)d_in[3];
  const float* mem_ts   = (const float*)d_in[4];
  const int*   mem_init = (const int*)  d_in[5];

  char* ws = (char*)d_ws;
  float*    acc = (float*)ws;                                     // 4 scalars
  _Float16* xh  = (_Float16*)(ws + 256);                          // B x 384 f16
  _Float16* mh  = (_Float16*)(ws + 256 + (size_t)B_SZ * DD * 2);  // C x 384 f16
  float*    wvp = (float*)(ws + 256 + (size_t)(B_SZ + C_SZ) * DD * 2);
  int*      inn = (int*)((char*)wvp + (size_t)C_SZ * 4);

  hipLaunchKernelGGL(k_init, dim3(1), dim3(32), 0, stream, acc);
  hipLaunchKernelGGL(k_update_mem, dim3(C_SZ), dim3(128), 0, stream,
                     y_true, y_pred, mem_emb, mem_ts, mem_init, mh, wvp, inn, acc);
  hipLaunchKernelGGL(k_norm_x, dim3(B_SZ), dim3(128), 0, stream,
                     y_true, y_pred, xh, acc);
  hipLaunchKernelGGL(k_pair_loss, dim3(528), dim3(256), 0, stream, xh, y_true, acc);
  hipLaunchKernelGGL(k_mem_loss, dim3(32, 32), dim3(256), 0, stream,
                     xh, mh, y_true, lookup, wvp, inn, acc);
  hipLaunchKernelGGL(k_final, dim3(1), dim3(32), 0, stream, acc, (float*)d_out);
}
